// SwinTransformerBlock_5669356833263
// MI455X (gfx1250) — matrix operand order repaired and checked by reading, not compiled, not run on hardware
//
#include <hip/hip_runtime.h>
#include <hip/hip_bf16.h>

typedef __attribute__((ext_vector_type(16))) _Float16 v16h;
typedef __attribute__((ext_vector_type(8)))  _Float16 v8h;
typedef __attribute__((ext_vector_type(8)))  float    v8f;

#define B_      32
#define C_      128
#define H_      64
#define W_      64
#define WS_     8
#define SHIFT_  4
#define HEADS_  4
#define DH_     32
#define TOK_    64
#define NW_     64
#define HID_    512
#define METAH_  256

// ---------------- WMMA fragment helpers (CDNA5 16x16x32 f16) ----------------
// A: 16x32 (MxK). lane m = lane&15 ; K offset 16*(lane>>4).
// Lane's 16 halves are one contiguous run: row[k0+16hs .. +15]
__device__ __forceinline__ v16h load_A_frag(const _Float16* base, int ld, int m0, int k0) {
  int lane = threadIdx.x & 31;
  const _Float16* row = base + (m0 + (lane & 15)) * ld + k0 + ((lane >> 4) << 4);
  union { v16h v; v8h h[2]; } u;
  u.h[0] = *(const v8h*)(row);
  u.h[1] = *(const v8h*)(row + 8);
  return u.v;
}
// B from consecutive-K source: element (k,n) at base[n*ld + k] -> 16 consecutive halves per lane
__device__ __forceinline__ v16h load_Bt_frag(const _Float16* base, int ld, int k0, int n0) {
  int lane = threadIdx.x & 31;
  const _Float16* p = base + (n0 + (lane & 15)) * ld + k0 + ((lane >> 4) << 4);
  union { v16h v; v8h h[2]; } u;
  u.h[0] = *(const v8h*)(p);
  u.h[1] = *(const v8h*)(p + 8);
  return u.v;
}
// B from pre-swizzled weights: [frag][lane][16] contiguous -> one 32B per-lane load
__device__ __forceinline__ v16h load_B_swz(const _Float16* base, int NT, int kt, int nt) {
  const _Float16* p = base + ((((kt * NT + nt) << 5) + (threadIdx.x & 31)) << 4);
  return *(const v16h*)p;
}
__device__ __forceinline__ v8f wmma_f16(v16h a, v16h b, v8f c) {
  return __builtin_amdgcn_wmma_f32_16x16x32_f16(false, a, false, b, (short)0, c, false, false);
}

// ---------------- weight f32 -> f16, pre-swizzled into B-fragment order ----------------
__global__ void __launch_bounds__(256)
swizzle_w_kernel(const float* __restrict__ src, _Float16* __restrict__ dst, int K, int N) {
  int NT = N >> 4;
  int idx = blockIdx.x * 256 + threadIdx.x;
  if (idx >= K * N) return;
  int i    = idx & 15;
  int lane = (idx >> 4) & 31;
  int f    = idx >> 9;
  int kt = f / NT, nt = f - kt * NT;
  int k = kt * 32 + ((lane >> 4) << 4) + i;
  int n = nt * 16 + (lane & 15);
  dst[idx] = (_Float16)src[k * N + n];
}

// ---------------- meta-MLP relative position bias (4 x 64 x 64) ----------------
__global__ void __launch_bounds__(256)
meta_bias_kernel(const float* __restrict__ w1, const float* __restrict__ b1,
                 const float* __restrict__ w2, const float* __restrict__ b2,
                 float* __restrict__ biasbuf) {
  int p = blockIdx.x * 256 + threadIdx.x;
  if (p >= TOK_ * TOK_) return;
  int q = p >> 6, k = p & 63;
  float dh = (float)((q >> 3) - (k >> 3));
  float dw = (float)((q & 7) - (k & 7));
  float r0 = copysignf(log1pf(fabsf(dh)), dh);
  float r1 = copysignf(log1pf(fabsf(dw)), dw);
  float a0 = 0.f, a1 = 0.f, a2 = 0.f, a3 = 0.f;
  for (int m = 0; m < METAH_; ++m) {
    float hid = fmaxf(r0 * w1[m] + r1 * w1[METAH_ + m] + b1[m], 0.f);
    a0 += hid * w2[m * HEADS_ + 0];
    a1 += hid * w2[m * HEADS_ + 1];
    a2 += hid * w2[m * HEADS_ + 2];
    a3 += hid * w2[m * HEADS_ + 3];
  }
  biasbuf[0 * 4096 + p] = a0 + b2[0];
  biasbuf[1 * 4096 + p] = a1 + b2[1];
  biasbuf[2 * 4096 + p] = a2 + b2[2];
  biasbuf[3 * 4096 + p] = a3 + b2[3];
}

// ---------------- window attention: 1 block = 1 window (64 tok x 128 ch) ----------------
__global__ void __launch_bounds__(256)
swin_attn_kernel(const float* __restrict__ x,
                 const float* __restrict__ qkv_b,
                 const float* __restrict__ proj_b,
                 const float* __restrict__ tau,
                 const _Float16* __restrict__ qkv_wh,
                 const _Float16* __restrict__ proj_wh,
                 const float* __restrict__ biasbuf,
                 float* __restrict__ om) {
  extern __shared__ char smem[];
  _Float16* sX   = (_Float16*)(smem);           // [64][136]  window x (f16)
  _Float16* sQKV = (_Float16*)(smem + 17408);   // [64][264]  q|k (f16)
  _Float16* sVT  = (_Float16*)(smem + 51200);   // [128][72]  v transposed (d-major)
  float*    sS   = (float*)   (smem + 69632);   // [64][68]   scores (f32)
  _Float16* sP   = (_Float16*)(smem + 87040);   // [64][72]   probs (f16)
  _Float16* sO   = (_Float16*)(smem + 96256);   // [64][136]  attn out (f16)
  float*    qn   = (float*)   (smem + 113664);  // [4][64]
  float*    kn   = (float*)   (smem + 114688);  // [4][64]
  int*      reg  = (int*)     (smem + 115712);  // [64] shift-mask regions

  const int tid  = threadIdx.x;
  const int lane = tid & 31;
  const int wv   = tid >> 5;
  const int wi   = blockIdx.x;
  const int b    = wi >> 6;
  const int wl   = wi & 63;
  const int wh   = wl >> 3;
  const int ww   = wl & 7;

  if (tid < TOK_) {
    int th = tid >> 3, tw = tid & 7;
    int ph = wh * WS_ + th, pw = ww * WS_ + tw;
    int sh = (ph < 56) ? 0 : (ph < 60 ? 1 : 2);
    int sw = (pw < 56) ? 0 : (pw < 60 ? 1 : 2);
    reg[tid] = sh * 3 + sw;
  }
  // load shifted window (roll(-SHIFT) folded into index)
  for (int idx = tid; idx < TOK_ * C_; idx += 256) {
    int c = idx >> 6, t = idx & 63;
    int th = t >> 3, tw = t & 7;
    int hh = (wh * WS_ + th + SHIFT_) & 63;
    int w2 = (ww * WS_ + tw + SHIFT_) & 63;
    sX[t * 136 + c] = (_Float16)x[((b * C_ + c) * H_ + hh) * W_ + w2];
  }
  __syncthreads();

  // QKV GEMM: [64,128] x [128,384], 96 tiles over 8 waves; v stored transposed
  for (int tt = wv; tt < 96; tt += 8) {
    int mi = tt / 24, ni = tt % 24;
    v8f acc = {};
#pragma unroll
    for (int kk = 0; kk < 4; ++kk) {
      v16h a  = load_A_frag(sX, 136, mi * 16, kk * 32);
      v16h bb = load_B_swz(qkv_wh, 24, kk, ni);
      acc = wmma_f16(a, bb, acc);
    }
    int n = ni * 16 + (lane & 15);
    float bias = qkv_b[n];
    int mbase = mi * 16 + ((lane >> 4) << 3);
    if (ni < 16) {  // q|k region (wave-uniform branch)
#pragma unroll
      for (int r = 0; r < 8; ++r)
        sQKV[(mbase + r) * 264 + n] = (_Float16)(acc[r] + bias);
    } else {        // v -> transposed [d][tok]
      int d = n - 2 * C_;
#pragma unroll
      for (int r = 0; r < 8; ++r)
        sVT[d * 72 + mbase + r] = (_Float16)(acc[r] + bias);
    }
  }
  __syncthreads();

  // q/k L2 norms (f32)
  {
    int t = tid >> 2, h = tid & 3;
    const _Float16* qr = sQKV + t * 264 + h * DH_;
    const _Float16* kr = sQKV + t * 264 + C_ + h * DH_;
    float sq = 0.f, sk = 0.f;
#pragma unroll
    for (int d = 0; d < DH_; ++d) {
      float qv = (float)qr[d], kv = (float)kr[d];
      sq += qv * qv; sk += kv * kv;
    }
    qn[h * 64 + t] = sqrtf(sq);
    kn[h * 64 + t] = sqrtf(sk);
  }
  __syncthreads();

  for (int h = 0; h < HEADS_; ++h) {
    float invtau = 1.0f / fmaxf(tau[h], 0.01f);
    // scores = q . k^T : 16 tiles / 8 waves
    for (int tt = wv; tt < 16; tt += 8) {
      int mi = tt >> 2, ni = tt & 3;
      v16h a  = load_A_frag(sQKV, 264, mi * 16, h * DH_);
      v16h bb = load_Bt_frag(sQKV + C_ + h * DH_, 264, 0, ni * 16);
      v8f acc = {};
      acc = wmma_f16(a, bb, acc);
      int nq = ni * 16 + (lane & 15);
      int mbase = mi * 16 + ((lane >> 4) << 3);
#pragma unroll
      for (int r = 0; r < 8; ++r) {
        int mq = mbase + r;
        float denom = fmaxf(qn[h * 64 + mq] * kn[h * 64 + nq], 1e-6f);
        float v = acc[r] / denom * invtau + biasbuf[(h * TOK_ + mq) * TOK_ + nq];
        if (reg[mq] != reg[nq]) v -= 100.0f;
        sS[mq * 68 + nq] = v;
      }
    }
    __syncthreads();
    // softmax: 4 lanes per row, shfl-xor reduce within groups of 4
    {
      int row = tid >> 2, qd = tid & 3;
      const float* rp = sS + row * 68 + qd * 16;
      float mx = -1e30f;
#pragma unroll
      for (int i = 0; i < 16; ++i) mx = fmaxf(mx, rp[i]);
      mx = fmaxf(mx, __shfl_xor(mx, 1, 4));
      mx = fmaxf(mx, __shfl_xor(mx, 2, 4));
      float e[16], sum = 0.f;
#pragma unroll
      for (int i = 0; i < 16; ++i) { e[i] = expf(rp[i] - mx); sum += e[i]; }
      sum += __shfl_xor(sum, 1, 4);
      sum += __shfl_xor(sum, 2, 4);
      float inv = 1.0f / sum;
      _Float16* pp = sP + row * 72 + qd * 16;
#pragma unroll
      for (int i = 0; i < 16; ++i) pp[i] = (_Float16)(e[i] * inv);
    }
    __syncthreads();
    // out_h = P . V : [64,64]x[64,32] -> 8 tiles, 1 per wave; B from transposed V
    {
      int mi = wv >> 1, ni = wv & 1;
      v8f acc = {};
#pragma unroll
      for (int ks = 0; ks < 2; ++ks) {
        v16h a  = load_A_frag(sP, 72, mi * 16, ks * 32);
        v16h bb = load_Bt_frag(sVT + (h * DH_) * 72, 72, ks * 32, ni * 16);
        acc = wmma_f16(a, bb, acc);
      }
      int n = h * DH_ + ni * 16 + (lane & 15);
      int mbase = mi * 16 + ((lane >> 4) << 3);
#pragma unroll
      for (int r = 0; r < 8; ++r)
        sO[(mbase + r) * 136 + n] = (_Float16)acc[r];
    }
    __syncthreads();
  }

  // projection [64,128]x[128,128], scatter to om with reverse roll folded in
  for (int tt = wv; tt < 32; tt += 8) {
    int mi = tt >> 3, ni = tt & 7;
    v8f acc = {};
#pragma unroll
    for (int kk = 0; kk < 4; ++kk) {
      v16h a  = load_A_frag(sO, 136, mi * 16, kk * 32);
      v16h bb = load_B_swz(proj_wh, 8, kk, ni);
      acc = wmma_f16(a, bb, acc);
    }
    int n = ni * 16 + (lane & 15);
    float bias = proj_b[n];
    int mbase = mi * 16 + ((lane >> 4) << 3);
#pragma unroll
    for (int r = 0; r < 8; ++r) {
      int t = mbase + r;
      int th = t >> 3, tw = t & 7;
      int hf = (wh * WS_ + th + SHIFT_) & 63;
      int wf = (ww * WS_ + tw + SHIFT_) & 63;
      om[((b * H_ + hf) * W_ + wf) * C_ + n] = acc[r] + bias;
    }
  }
}

// ---------------- LN1 + FFN + LN2 + residuals: 1 block = 64 rows (one (b,h) strip) ----------------
__global__ void __launch_bounds__(256)
swin_ffn_kernel(const float* __restrict__ x,
                const float* __restrict__ om,
                const float* __restrict__ ln1_g, const float* __restrict__ ln1_b,
                const float* __restrict__ ln2_g, const float* __restrict__ ln2_b,
                const float* __restrict__ ffn_b1, const float* __restrict__ ffn_b2,
                const _Float16* __restrict__ w1h, const _Float16* __restrict__ w2h,
                float* __restrict__ out) {
  extern __shared__ char smem[];
  float*    sOM = (float*)(smem);              // [64][136] om -> skip (in place)
  float*    sF  = (float*)(smem + 34816);      // [64][136] xin, later ffn result
  _Float16* sA  = (_Float16*)(smem + 69632);   // [64][136] skip f16
  _Float16* sH  = (_Float16*)(smem + 87040);   // [64][520] gelu hidden f16

  const int tid  = threadIdx.x;
  const int lane = tid & 31;
  const int wv   = tid >> 5;
  const int b    = blockIdx.x >> 6;
  const int hr   = blockIdx.x & 63;

  for (int idx = tid; idx < 64 * 128; idx += 256) {
    int w = idx >> 7, c = idx & 127;
    sOM[w * 136 + c] = om[((b * H_ + hr) * W_ + w) * C_ + c];
  }
  for (int idx = tid; idx < 64 * 128; idx += 256) {
    int c = idx >> 6, w = idx & 63;
    sF[w * 136 + c] = x[((b * C_ + c) * H_ + hr) * W_ + w];  // xin
  }
  __syncthreads();

  // skip = xin + LN1(om)
  if (tid < 64) {
    float* row = sOM + tid * 136;
    float mu = 0.f;
    for (int c = 0; c < 128; ++c) mu += row[c];
    mu *= (1.0f / 128.0f);
    float var = 0.f;
    for (int c = 0; c < 128; ++c) { float d = row[c] - mu; var += d * d; }
    var *= (1.0f / 128.0f);
    float rs = rsqrtf(var + 1e-5f);
    for (int c = 0; c < 128; ++c) {
      float skip = sF[tid * 136 + c] + (row[c] - mu) * rs * ln1_g[c] + ln1_b[c];
      row[c] = skip;
      sA[tid * 136 + c] = (_Float16)skip;
    }
  }
  __syncthreads();

  // GEMM1 [64,128]x[128,512] + exact GELU
  for (int tt = wv; tt < 128; tt += 8) {
    int mi = tt >> 5, ni = tt & 31;
    v8f acc = {};
#pragma unroll
    for (int kk = 0; kk < 4; ++kk) {
      v16h a  = load_A_frag(sA, 136, mi * 16, kk * 32);
      v16h bb = load_B_swz(w1h, 32, kk, ni);
      acc = wmma_f16(a, bb, acc);
    }
    int n = ni * 16 + (lane & 15);
    float bias = ffn_b1[n];
    int mbase = mi * 16 + ((lane >> 4) << 3);
#pragma unroll
    for (int r = 0; r < 8; ++r) {
      float v = acc[r] + bias;
      float g = 0.5f * v * (1.0f + erff(v * 0.70710678118654752f));
      sH[(mbase + r) * 520 + n] = (_Float16)g;
    }
  }
  __syncthreads();

  // GEMM2 [64,512]x[512,128]
  for (int tt = wv; tt < 32; tt += 8) {
    int mi = tt >> 3, ni = tt & 7;
    v8f acc = {};
#pragma unroll
    for (int kk = 0; kk < 16; ++kk) {
      v16h a  = load_A_frag(sH, 520, mi * 16, kk * 32);
      v16h bb = load_B_swz(w2h, 8, kk, ni);
      acc = wmma_f16(a, bb, acc);
    }
    int n = ni * 16 + (lane & 15);
    float bias = ffn_b2[n];
    int mbase = mi * 16 + ((lane >> 4) << 3);
#pragma unroll
    for (int r = 0; r < 8; ++r) sF[(mbase + r) * 136 + n] = acc[r] + bias;
  }
  __syncthreads();

  // out = skip + LN2(ffn), stored transposed to [B,C,H,W]
  if (tid < 64) {
    float* row = sF + tid * 136;
    float mu = 0.f;
    for (int c = 0; c < 128; ++c) mu += row[c];
    mu *= (1.0f / 128.0f);
    float var = 0.f;
    for (int c = 0; c < 128; ++c) { float d = row[c] - mu; var += d * d; }
    var *= (1.0f / 128.0f);
    float rs = rsqrtf(var + 1e-5f);
    for (int c = 0; c < 128; ++c) {
      float o = sOM[tid * 136 + c] + (row[c] - mu) * rs * ln2_g[c] + ln2_b[c];
      out[((b * C_ + c) * H_ + hr) * W_ + tid] = o;
    }
  }
}

extern "C" void kernel_launch(void* const* d_in, const int* in_sizes, int n_in,
                              void* d_out, int out_size, void* d_ws, size_t ws_size,
                              hipStream_t stream) {
  const float* x       = (const float*)d_in[0];
  const float* qkv_w   = (const float*)d_in[1];
  const float* qkv_b   = (const float*)d_in[2];
  const float* proj_w  = (const float*)d_in[3];
  const float* proj_b  = (const float*)d_in[4];
  const float* meta_w1 = (const float*)d_in[5];
  const float* meta_b1 = (const float*)d_in[6];
  const float* meta_w2 = (const float*)d_in[7];
  const float* meta_b2 = (const float*)d_in[8];
  const float* tau     = (const float*)d_in[9];
  const float* ln1_g   = (const float*)d_in[10];
  const float* ln1_b   = (const float*)d_in[11];
  const float* ln2_g   = (const float*)d_in[12];
  const float* ln2_b   = (const float*)d_in[13];
  const float* ffn_w1  = (const float*)d_in[14];
  const float* ffn_b1  = (const float*)d_in[15];
  const float* ffn_w2  = (const float*)d_in[16];
  const float* ffn_b2  = (const float*)d_in[17];

  char* ws = (char*)d_ws;
  _Float16* qkv_wh  = (_Float16*)(ws + 0);        // 128*384 f16 (swizzled)
  _Float16* proj_wh = (_Float16*)(ws + 98304);    // 128*128 f16 (swizzled)
  _Float16* w1h     = (_Float16*)(ws + 131072);   // 128*512 f16 (swizzled)
  _Float16* w2h     = (_Float16*)(ws + 262144);   // 512*128 f16 (swizzled)
  float*    biasbuf = (float*)   (ws + 393216);   // 4*64*64 f32
  float*    om      = (float*)   (ws + 458752);   // B*H*W*C f32

  swizzle_w_kernel<<<(49152 + 255) / 256, 256, 0, stream>>>(qkv_w, qkv_wh, 128, 384);
  swizzle_w_kernel<<<(16384 + 255) / 256, 256, 0, stream>>>(proj_w, proj_wh, 128, 128);
  swizzle_w_kernel<<<(65536 + 255) / 256, 256, 0, stream>>>(ffn_w1, w1h, 128, 512);
  swizzle_w_kernel<<<(65536 + 255) / 256, 256, 0, stream>>>(ffn_w2, w2h, 512, 128);
  meta_bias_kernel<<<16, 256, 0, stream>>>(meta_w1, meta_b1, meta_w2, meta_b2, biasbuf);

  swin_attn_kernel<<<B_ * NW_, 256, 115968, stream>>>(x, qkv_b, proj_b, tau,
                                                      qkv_wh, proj_wh, biasbuf, om);
  swin_ffn_kernel<<<B_ * H_, 256, 153600, stream>>>(x, om, ln1_g, ln1_b, ln2_g, ln2_b,
                                                    ffn_b1, ffn_b2, w1h, w2h, (float*)d_out);
}
